// Dynamics_35227321762341
// MI455X (gfx1250) — matrix operand order repaired and checked by reading, not compiled, not run on hardware
//
#include <hip/hip_runtime.h>
#include <hip/hip_bf16.h>
#include <math.h>
#include <stdint.h>

#define IN_DIM   128
#define HID      2048
#define OUT_DIM  128
#define HSTEPS   1024
#define GATES    (4*HID)        // 8192
#define NBLK     64             // persistent blocks (hidden slices)
#define JPB      (HID/NBLK)     // 32 hidden units per block
#define TPB      256            // 8 waves

#define LOG2PI_F 1.8378770664093453f

typedef __attribute__((ext_vector_type(8)))  __bf16 v8bf;
typedef __attribute__((ext_vector_type(16))) __bf16 v16bf;
typedef __attribute__((ext_vector_type(8)))  float  v8f;
typedef __attribute__((ext_vector_type(4)))  int    v4i;

#if __has_builtin(__builtin_amdgcn_global_load_async_to_lds_b128) && \
    __has_builtin(__builtin_amdgcn_s_wait_asynccnt)
#define USE_ASYNC_LDS 1
// pointer-to-(AS1 v4i) and pointer-to-(AS3 v4i)
typedef __attribute__((address_space(1))) v4i* gv4i_ptr;
typedef __attribute__((address_space(3))) v4i* lv4i_ptr;
#endif

// ---- workspace layout (bytes, 256-aligned regions) ----
#define OFF_CNT   0u                                   // 2 x u32 barrier state
#define OFF_HBUF  256u                                 // 2*HID bf16 double buffer
#define OFF_S0B   (256u + 8192u)                       // s0 bf16: H*IN*2 = 262144
#define OFF_WIHB  (OFF_S0B  + 262144u)                 // W_ih bf16: 2 MB
#define OFF_WHHB  (OFF_WIHB + 2097152u)                // W_hh bf16: 32 MB
#define OFF_XP    (OFF_WHHB + 33554432u)               // XP f32: H*GATES*4 = 32 MB
// total ~66.3 MB

// ---------------- device-wide barrier (sense via generation counter) --------
__device__ __forceinline__ void grid_sync(unsigned* cnt, unsigned* gen) {
  __threadfence();                     // make this block's global writes visible
  __syncthreads();
  if (threadIdx.x == 0) {
    unsigned g = __hip_atomic_load(gen, __ATOMIC_RELAXED, __HIP_MEMORY_SCOPE_AGENT);
    unsigned a = __hip_atomic_fetch_add(cnt, 1u, __ATOMIC_ACQ_REL, __HIP_MEMORY_SCOPE_AGENT);
    if (a == NBLK - 1) {
      __hip_atomic_store(cnt, 0u, __ATOMIC_RELAXED, __HIP_MEMORY_SCOPE_AGENT);
      __hip_atomic_fetch_add(gen, 1u, __ATOMIC_RELEASE, __HIP_MEMORY_SCOPE_AGENT);
    } else {
      while (__hip_atomic_load(gen, __ATOMIC_ACQUIRE, __HIP_MEMORY_SCOPE_AGENT) == g)
        __builtin_amdgcn_s_sleep(1);
    }
  }
  __syncthreads();
  __threadfence();                     // acquire other blocks' writes
}

// ---------------- kernel 1: fp32 -> bf16 weight/activation staging ----------
__global__ void prep_convert(const float* __restrict__ s0,
                             const float* __restrict__ W_ih,
                             const float* __restrict__ W_hh,
                             char* __restrict__ ws) {
  __bf16* s0b  = (__bf16*)(ws + OFF_S0B);
  __bf16* wihb = (__bf16*)(ws + OFF_WIHB);
  __bf16* whhb = (__bf16*)(ws + OFF_WHHB);
  __bf16* hbuf = (__bf16*)(ws + OFF_HBUF);
  unsigned* cnt = (unsigned*)(ws + OFF_CNT);
  long stride = (long)gridDim.x * blockDim.x;
  long gid = (long)blockIdx.x * blockDim.x + threadIdx.x;
  for (long i = gid; i < (long)GATES * HID;      i += stride) whhb[i] = (__bf16)W_hh[i];
  for (long i = gid; i < (long)GATES * IN_DIM;   i += stride) wihb[i] = (__bf16)W_ih[i];
  for (long i = gid; i < (long)HSTEPS * IN_DIM;  i += stride) s0b[i]  = (__bf16)s0[i];
  for (long i = gid; i < 2 * HID;                i += stride) hbuf[i] = (__bf16)0.0f;
  if (gid < 2) cnt[gid] = 0u;        // reset barrier count + generation
}

// ---------------- kernel 2: XP[t, r] = W_ih @ x_t + b_ih + b_hh (WMMA GEMM) --
// M = time (1024), N = gate rows (8192), K = 128. One 16x16 tile per wave.
__global__ void xp_gemm(const float* __restrict__ b_ih,
                        const float* __restrict__ b_hh,
                        char* __restrict__ ws) {
  const __bf16* s0b  = (const __bf16*)(ws + OFF_S0B);
  const __bf16* wihb = (const __bf16*)(ws + OFF_WIHB);
  float* XP = (float*)(ws + OFF_XP);

  int lane = threadIdx.x & 31;
  int wave = threadIdx.x >> 5;
  int tile = blockIdx.x * 8 + wave;        // 32768 tiles total
  int tileN = tile & 511;                  // GATES/16
  int tileM = tile >> 9;                   // HSTEPS/16
  int hiHalf = (lane >= 16);
  int mrow = tileM * 16 + (lane & 15);     // A row (time)
  int nrow = tileN * 16 + (lane & 15);     // B column (gate row of W_ih)

  v8f acc = {};
  #pragma unroll
  for (int kk = 0; kk < IN_DIM / 32; ++kk) {
    int kb = kk * 32;
    // A (16x32 bf16): lanes 0-15 carry K {0..7,16..23}, lanes 16-31 {8..15,24..31}
    const __bf16* pa = s0b + (long)mrow * IN_DIM + kb + (hiHalf ? 8 : 0);
    v8bf alo = *(const v8bf*)pa;
    v8bf ahi = *(const v8bf*)(pa + 16);
    v16bf a = __builtin_shufflevector(alo, ahi, 0,1,2,3,4,5,6,7,8,9,10,11,12,13,14,15);
    // B (32x16 bf16) = W_ih^T tile: lanes 0-15 carry K {0..7,16..23}, lanes 16-31 {8..15,24..31}
    const __bf16* pb = wihb + (long)nrow * IN_DIM + kb + (hiHalf ? 8 : 0);
    v16bf b = __builtin_shufflevector(*(const v8bf*)pb, *(const v8bf*)(pb + 16), 0,1,2,3,4,5,6,7,8,9,10,11,12,13,14,15);
    acc = __builtin_amdgcn_wmma_f32_16x16x32_bf16(false, a, false, b,
                                                  (short)0, acc, false, false);
  }
  float bias = b_ih[nrow] + b_hh[nrow];
  int tbase = tileM * 16 + (hiHalf ? 8 : 0);   // C: VGPR r -> M=r (lanes<16) / 8+r
  #pragma unroll
  for (int r = 0; r < 8; ++r)
    XP[(long)(tbase + r) * GATES + nrow] = acc[r] + bias;
}

// ---------------- kernel 3: persistent LSTM recurrence ----------------------
// Block b owns hidden slice j in [b*32, b*32+32): 128 gate rows (4 gates x 32)
// as 8 16-row WMMA tiles (one per wave), plus output dims {2b, 2b+1}.
__global__ void __launch_bounds__(TPB, 1)
lstm_persistent(const float* __restrict__ eps,
                const float* __restrict__ W_mu, const float* __restrict__ b_mu,
                const float* __restrict__ W_lv, const float* __restrict__ b_lv,
                float* __restrict__ out, char* __restrict__ ws) {
  const __bf16* whhb = (const __bf16*)(ws + OFF_WHHB);
  const float*  XP   = (const float*) (ws + OFF_XP);
  __bf16* hbuf = (__bf16*)(ws + OFF_HBUF);
  unsigned* cnt = (unsigned*)(ws + OFF_CNT);
  unsigned* gen = cnt + 1;

  float* xs  = out;
  float* lps = out + (long)HSTEPS * OUT_DIM;
  float* hs  = out + 2l * HSTEPS * OUT_DIM;

  __shared__ __align__(32) __bf16 h_s[HID];   // per-step h broadcast (bf16, 4 KB)
  __shared__ float gs[4][JPB];     // raw gate pre-activations for this slice
  __shared__ float c_s[JPB];       // persistent cell state slice
  __shared__ float red[4][TPB];    // head reduction

  const int tid  = threadIdx.x;
  const int lane = tid & 31;
  const int wave = tid >> 5;
  const int blk  = blockIdx.x;
  const int g    = wave >> 1;          // gate index 0..3 (i,f,g,o)
  const int jt   = wave & 1;           // 16-row sub-tile within the 32-j slice
  const long row_base = (long)g * HID + blk * JPB + jt * 16;
  const int  hiHalf   = (lane >= 16);
  const __bf16* wrow = whhb + (row_base + (lane & 15)) * (long)HID + (hiHalf ? 8 : 0);
  const __bf16* hlds = h_s + (hiHalf ? 8 : 0);

  const int d0 = blk * 2;
  const float* wmu0 = W_mu + (long)d0 * HID;
  const float* wmu1 = wmu0 + HID;
  const float* wlv0 = W_lv + (long)d0 * HID;
  const float* wlv1 = wlv0 + HID;

  if (tid < JPB) c_s[tid] = 0.0f;
  __syncthreads();

  for (int t = 0; t < HSTEPS; ++t) {
    const __bf16* hin = hbuf + (t & 1) * HID;
    __bf16* hout = hbuf + ((t + 1) & 1) * HID;

    // ---- stage h_t-1 into LDS: one 16-byte chunk per thread ----------------
#ifdef USE_ASYNC_LDS
    {
      gv4i_ptr gp = (gv4i_ptr)(uintptr_t)(hin + tid * 8);           // AS1, 64-bit
      lv4i_ptr lp = (lv4i_ptr)(unsigned)(uintptr_t)(h_s + tid * 8); // AS3, 32-bit offset
      __builtin_amdgcn_global_load_async_to_lds_b128(gp, lp, 0, 0);
      __builtin_amdgcn_s_wait_asynccnt(0);
    }
#else
    *(v8bf*)(h_s + tid * 8) = *(const v8bf*)(hin + tid * 8);
#endif
    __syncthreads();

    // ---- gates slice: 16 rows x 2048-K bf16 matvec via WMMA broadcast-B ----
    // Two independent accumulators break the WMMA D->C dependency chain.
    v8f acc0 = {}, acc1 = {};
    #pragma unroll 4
    for (int kk = 0; kk < HID / 32; kk += 2) {
      {
        int kb = kk * 32;
        const __bf16* pa = wrow + kb;
        v8bf alo = *(const v8bf*)pa;
        v8bf ahi = *(const v8bf*)(pa + 16);
        v16bf a = __builtin_shufflevector(alo, ahi,
                    0,1,2,3,4,5,6,7,8,9,10,11,12,13,14,15);
        v16bf b = __builtin_shufflevector(*(const v8bf*)(hlds + kb), *(const v8bf*)(hlds + kb + 16), 0,1,2,3,4,5,6,7,8,9,10,11,12,13,14,15);        // LDS broadcast
        acc0 = __builtin_amdgcn_wmma_f32_16x16x32_bf16(false, a, false, b,
                                                       (short)0, acc0, false, false);
        if (kk + 16 < HID / 32)
          __builtin_prefetch((const void*)(pa + 16 * 32), 0, 1); // global_prefetch_b8
      }
      {
        int kb = (kk + 1) * 32;
        const __bf16* pa = wrow + kb;
        v8bf alo = *(const v8bf*)pa;
        v8bf ahi = *(const v8bf*)(pa + 16);
        v16bf a = __builtin_shufflevector(alo, ahi,
                    0,1,2,3,4,5,6,7,8,9,10,11,12,13,14,15);
        v16bf b = __builtin_shufflevector(*(const v8bf*)(hlds + kb), *(const v8bf*)(hlds + kb + 16), 0,1,2,3,4,5,6,7,8,9,10,11,12,13,14,15);
        acc1 = __builtin_amdgcn_wmma_f32_16x16x32_bf16(false, a, false, b,
                                                       (short)0, acc1, false, false);
      }
    }
    // C column group: lane 0 holds M=0..7, lane 16 holds M=8..15
    const float* xp_t = XP + (long)t * GATES;
    if (lane == 0) {
      #pragma unroll
      for (int r = 0; r < 8; ++r)
        gs[g][jt * 16 + r] = acc0[r] + acc1[r] + xp_t[row_base + r];
    } else if (lane == 16) {
      #pragma unroll
      for (int r = 0; r < 8; ++r)
        gs[g][jt * 16 + 8 + r] = acc0[r] + acc1[r] + xp_t[row_base + 8 + r];
    }
    __syncthreads();

    // ---- activations + state update for the 32 owned hidden units ---------
    if (tid < JPB) {
      int j = tid;
      float iv = 1.0f / (1.0f + __expf(-gs[0][j]));
      float fv = 1.0f / (1.0f + __expf(-gs[1][j]));
      float gv = tanhf(gs[2][j]);
      float ov = 1.0f / (1.0f + __expf(-gs[3][j]));
      float c  = fv * c_s[j] + iv * gv;
      c_s[j] = c;
      float h = ov * tanhf(c);
      int jg = blk * JPB + j;
      hs[(long)t * HID + jg] = h;        // f32 output (also head input)
      hout[jg] = (__bf16)h;              // bf16 broadcast for next matvec
    }

    grid_sync(cnt, gen);                 // h_t fully published device-wide

    // ---- heads for step t: dims d0, d0+1 (mu & logvar) ---------------------
    const float* hrow = hs + (long)t * HID;
    float p0 = 0.f, p1 = 0.f, p2 = 0.f, p3 = 0.f;
    for (int k = tid; k < HID; k += TPB) {
      float hv = hrow[k];
      p0 += wmu0[k] * hv;  p1 += wmu1[k] * hv;
      p2 += wlv0[k] * hv;  p3 += wlv1[k] * hv;
    }
    red[0][tid] = p0; red[1][tid] = p1; red[2][tid] = p2; red[3][tid] = p3;
    __syncthreads();
    for (int s = TPB / 2; s > 0; s >>= 1) {
      if (tid < s) {
        red[0][tid] += red[0][tid + s];
        red[1][tid] += red[1][tid + s];
        red[2][tid] += red[2][tid + s];
        red[3][tid] += red[3][tid + s];
      }
      __syncthreads();
    }
    if (tid == 0) {
      float mu[2] = { red[0][0] + b_mu[d0], red[1][0] + b_mu[d0 + 1] };
      float lv[2] = { red[2][0] + b_lv[d0], red[3][0] + b_lv[d0 + 1] };
      #pragma unroll
      for (int q = 0; q < 2; ++q) {
        int d = d0 + q;
        float e = eps[(long)t * OUT_DIM + d];
        float stdv = __expf(0.5f * lv[q]);
        xs [(long)t * OUT_DIM + d] = mu[q] + stdv * e;
        // (x - mu)/std == e exactly, log(std) == 0.5*logvar
        lps[(long)t * OUT_DIM + d] = -0.5f * e * e - 0.5f * lv[q] - 0.5f * LOG2PI_F;
      }
    }
    __syncthreads();
  }
}

extern "C" void kernel_launch(void* const* d_in, const int* in_sizes, int n_in,
                              void* d_out, int out_size, void* d_ws, size_t ws_size,
                              hipStream_t stream) {
  (void)in_sizes; (void)n_in; (void)out_size; (void)ws_size;
  const float* s0   = (const float*)d_in[0];
  const float* eps  = (const float*)d_in[1];
  const float* W_ih = (const float*)d_in[2];
  const float* W_hh = (const float*)d_in[3];
  const float* b_ih = (const float*)d_in[4];
  const float* b_hh = (const float*)d_in[5];
  const float* W_mu = (const float*)d_in[6];
  const float* b_mu = (const float*)d_in[7];
  const float* W_lv = (const float*)d_in[8];
  const float* b_lv = (const float*)d_in[9];
  char*  ws  = (char*)d_ws;
  float* out = (float*)d_out;

  prep_convert<<<1024, 256, 0, stream>>>(s0, W_ih, W_hh, ws);
  xp_gemm<<<(HSTEPS / 16) * (GATES / 16) / 8, 256, 0, stream>>>(b_ih, b_hh, ws);
  lstm_persistent<<<NBLK, TPB, 0, stream>>>(eps, W_mu, b_mu, W_lv, b_lv, out, ws);
}
